// MultiHeadAttention_60507499266336
// MI455X (gfx1250) — compile-verified, matrix operand order checked by reading, not run on hardware
//
#include <hip/hip_runtime.h>

typedef __bf16 bf16;
typedef __attribute__((ext_vector_type(16))) __bf16 v16bf;
typedef __attribute__((ext_vector_type(8)))  __bf16 v8bf;
typedef __attribute__((ext_vector_type(4)))  __bf16 v4bf;
typedef __attribute__((ext_vector_type(8)))  float  v8f;

#define BB  4
#define NN  1024   // ND == NE
#define DD  1024
#define HH  8
#define DKK 128

// ---------------------------------------------------------------------------
// Fragment loader: row-major bf16 matrix, leading dim ld.
// A-operand layout (16-bit, 16x32): lanes 0-15 -> rows, elems 0..7 = K k0+0..7,
// elems 8..15 = K k0+16..23; lanes 16-31 same rows, K ranges +8.
// B-operand uses the identical layout sourced from the transposed matrix.
// ---------------------------------------------------------------------------
__device__ __forceinline__ v16bf load_frag(const bf16* __restrict__ base, int ld,
                                           int r0, int k0) {
    int lane = threadIdx.x & 31;
    int row  = r0 + (lane & 15);
    int kk   = k0 + ((lane >> 4) << 3);
    const bf16* p = base + (size_t)row * ld + kk;
    v8bf lo = *(const v8bf*)(p);        // 16B load
    v8bf hi = *(const v8bf*)(p + 16);   // 16B load
    return __builtin_shufflevector(lo, hi, 0,1,2,3,4,5,6,7,8,9,10,11,12,13,14,15);
}

__device__ __forceinline__ v8f wmma_bf16(v16bf a, v16bf b, v8f c) {
    return __builtin_amdgcn_wmma_f32_16x16x32_bf16(false, a, false, b,
                                                   (short)0, c, false, false);
}

// ---------------------------------------------------------------------------
// Prep: W[h][d][j] (f32) -> Wt[h*128+j][d] (bf16)   (grid=1024, block=256)
// ---------------------------------------------------------------------------
__global__ void k_transpose_w(const float* __restrict__ W, bf16* __restrict__ Wt) {
    int row = blockIdx.x;            // h*128 + j
    int h = row >> 7, j = row & 127;
    const float* src = W + (size_t)h * DD * DKK + j;
    bf16* dst = Wt + (size_t)row * DD;
    for (int d = threadIdx.x; d < DD; d += blockDim.x)
        dst[d] = (bf16)src[(size_t)d * DKK];
}

// w_vo[h][d] = sum_c W_V[h][d][c] * W_O[h*128+c]     (grid=32, block=256)
__global__ void k_wvo(const float* __restrict__ WV, const float* __restrict__ WO,
                      float* __restrict__ wvo) {
    int idx = blockIdx.x * blockDim.x + threadIdx.x;   // h*1024 + d
    int h = idx >> 10, d = idx & 1023;
    const float* wv = WV + ((size_t)h * DD + d) * DKK;
    const float* wo = WO + h * DKK;
    float s = 0.f;
#pragma unroll 8
    for (int c = 0; c < DKK; ++c) s += wv[c] * wo[c];
    wvo[idx] = s;
}

// f32 -> bf16 convert, 4 elems/thread                 (grid=4096, block=256)
__global__ void k_cvt(const float* __restrict__ x, bf16* __restrict__ y) {
    int i = (blockIdx.x * blockDim.x + threadIdx.x) * 4;
    float4 v = *(const float4*)(x + i);
    v4bf o;
    o[0] = (bf16)v.x; o[1] = (bf16)v.y; o[2] = (bf16)v.z; o[3] = (bf16)v.w;
    *(v4bf*)(y + i) = o;
}

// res_bias[r] = input_d[r,:] . W_O[:,0]               (grid=4096, block=256)
__global__ void k_resbias(const float* __restrict__ xd, const float* __restrict__ WO,
                          float* __restrict__ rb) {
    int r = blockIdx.x;
    __shared__ float red[8];
    const float* x = xd + (size_t)r * DD;
    float s = 0.f;
    for (int d = threadIdx.x; d < DD; d += 256) s += x[d] * WO[d];
    for (int off = 16; off; off >>= 1) s += __shfl_down(s, off, 32);
    if ((threadIdx.x & 31) == 0) red[threadIdx.x >> 5] = s;
    __syncthreads();
    if (threadIdx.x == 0) {
        float t = 0.f;
        for (int i = 0; i < 8; ++i) t += red[i];
        rb[r] = t;
    }
}

// vo[b][h][m] = input_e[b,m,:] . w_vo[h,:]            (grid=4096, block=256)
__global__ void k_vo(const float* __restrict__ xe, const float* __restrict__ wvo,
                     float* __restrict__ vo) {
    int r = blockIdx.x;                 // b*1024 + m
    __shared__ float xrow[DD];
    const float* x = xe + (size_t)r * DD;
    for (int d = threadIdx.x; d < DD; d += 256) xrow[d] = x[d];
    __syncthreads();
    int w = threadIdx.x >> 5, lane = threadIdx.x & 31;
    const float* wv = wvo + (size_t)w * DD;
    float s = 0.f;
    for (int d = lane; d < DD; d += 32) s += xrow[d] * wv[d];
    for (int off = 16; off; off >>= 1) s += __shfl_down(s, off, 32);
    if (lane == 0) {
        int b = r >> 10, m = r & 1023;
        vo[((size_t)b * HH + w) * NN + m] = s;
    }
}

// ---------------------------------------------------------------------------
// Projection GEMM: Out[4096x1024](bf16) = X[4096x1024] @ Wt^T
// block = 256 thr (8 waves, 2x4), block tile 128x128, wave tile 64x32.
// grid = dim3(8, 32)
// ---------------------------------------------------------------------------
__global__ __launch_bounds__(256) void k_gemm(const bf16* __restrict__ X,
                                              const bf16* __restrict__ Wt,
                                              bf16* __restrict__ Out) {
    int wid = threadIdx.x >> 5;
    int wm = wid >> 2, wn = wid & 3;
    int row0 = blockIdx.y * 128 + wm * 64;
    int col0 = blockIdx.x * 128 + wn * 32;
    v8f acc[4][2];
#pragma unroll
    for (int i = 0; i < 4; ++i)
#pragma unroll
        for (int j = 0; j < 2; ++j) acc[i][j] = v8f{};

    for (int k0 = 0; k0 < DD; k0 += 32) {
        v16bf a[4], b[2];
#pragma unroll
        for (int i = 0; i < 4; ++i) a[i] = load_frag(X, DD, row0 + i * 16, k0);
#pragma unroll
        for (int j = 0; j < 2; ++j) b[j] = load_frag(Wt, DD, col0 + j * 16, k0);
#pragma unroll
        for (int i = 0; i < 4; ++i)
#pragma unroll
            for (int j = 0; j < 2; ++j) acc[i][j] = wmma_bf16(a[i], b[j], acc[i][j]);
    }

    int lane = threadIdx.x & 31;
    int nc = lane & 15, half = lane >> 4;
#pragma unroll
    for (int i = 0; i < 4; ++i)
#pragma unroll
        for (int j = 0; j < 2; ++j)
#pragma unroll
            for (int r = 0; r < 8; ++r) {
                int rr = row0 + i * 16 + r + half * 8;
                int cc = col0 + j * 16 + nc;
                Out[(size_t)rr * DD + cc] = (bf16)acc[i][j][r];
            }
}

// ---------------------------------------------------------------------------
// Flash pass 1: per (b,h,n) row max + 1/sum(exp) over all m.
// block = 256 thr (8 waves), wave handles 16 rows. grid = dim3(8, 32).
// ---------------------------------------------------------------------------
__global__ __launch_bounds__(256) void k_stats(const bf16* __restrict__ Qp,
                                               const bf16* __restrict__ Kp,
                                               float* __restrict__ stats) {
    int bh = blockIdx.y;                 // b*8 + h
    int b = bh >> 3, h = bh & 7;
    int wid = threadIdx.x >> 5;
    int n0 = blockIdx.x * 128 + wid * 16;
    int lane = threadIdx.x & 31;
    const float scale = 1.0f / 128.0f;

    v16bf a[4];
#pragma unroll
    for (int j = 0; j < 4; ++j)
        a[j] = load_frag(Qp, DD, b * NN + n0, h * DKK + j * 32);

    float mx[8], sm[8];
#pragma unroll
    for (int r = 0; r < 8; ++r) { mx[r] = -1e30f; sm[r] = 0.f; }

    for (int m0 = 0; m0 < NN; m0 += 16) {
        v8f c = v8f{};
#pragma unroll
        for (int j = 0; j < 4; ++j) {
            v16bf bf = load_frag(Kp, DD, b * NN + m0, h * DKK + j * 32);
            c = wmma_bf16(a[j], bf, c);
        }
#pragma unroll
        for (int r = 0; r < 8; ++r) {
            float v = c[r] * scale;
            float nm = fmaxf(mx[r], v);
            sm[r] = sm[r] * __expf(mx[r] - nm) + __expf(v - nm);
            mx[r] = nm;
        }
    }
    // reduce (max,sum) across the 16 lanes of each half
#pragma unroll
    for (int r = 0; r < 8; ++r) {
        float m = mx[r], s = sm[r];
        for (int off = 1; off < 16; off <<= 1) {
            float om = __shfl_xor(m, off, 32);
            float os = __shfl_xor(s, off, 32);
            float nm = fmaxf(m, om);
            s = s * __expf(m - nm) + os * __expf(om - nm);
            m = nm;
        }
        if ((lane & 15) == 0) {
            int row = n0 + r + (lane >> 4) * 8;
            size_t idx = ((size_t)bh * NN + row) * 2;
            stats[idx + 0] = m;
            stats[idx + 1] = __builtin_amdgcn_rcpf(s);   // store inverse sum
        }
    }
}

// ---------------------------------------------------------------------------
// Pass 2: out[b,n,m] = sum_h softmax_h[n,m]*vo[b,h,m] + rb[b,n], * masks.
// block = 256 thr (8 waves, 2x4), block tile 64n x 128m, wave tile 32x32.
// grid = dim3(8, 16, 4).
// ---------------------------------------------------------------------------
__global__ __launch_bounds__(256) void k_out(const bf16* __restrict__ Qp,
                                             const bf16* __restrict__ Kp,
                                             const float* __restrict__ stats,
                                             const float* __restrict__ vo,
                                             const float* __restrict__ rb,
                                             const float* __restrict__ mask_d,
                                             const float* __restrict__ mask_e,
                                             float* __restrict__ out) {
    int b = blockIdx.z;
    int wid = threadIdx.x >> 5;
    int wm = wid >> 2, wn = wid & 3;
    int n0 = blockIdx.y * 64 + wm * 32;
    int m0 = blockIdx.x * 128 + wn * 32;
    int lane = threadIdx.x & 31;
    int nc = lane & 15, half = lane >> 4;
    const float scale = 1.0f / 128.0f;

    float acc[2][2][8];
#pragma unroll
    for (int ni = 0; ni < 2; ++ni)
#pragma unroll
        for (int mi = 0; mi < 2; ++mi)
#pragma unroll
            for (int r = 0; r < 8; ++r) acc[ni][mi][r] = 0.f;

    for (int h = 0; h < HH; ++h) {
        v8f c[2][2];
#pragma unroll
        for (int ni = 0; ni < 2; ++ni)
#pragma unroll
            for (int mi = 0; mi < 2; ++mi) c[ni][mi] = v8f{};

#pragma unroll
        for (int j = 0; j < 4; ++j) {
            v16bf a0 = load_frag(Qp, DD, b * NN + n0,      h * DKK + j * 32);
            v16bf a1 = load_frag(Qp, DD, b * NN + n0 + 16, h * DKK + j * 32);
            v16bf b0 = load_frag(Kp, DD, b * NN + m0,      h * DKK + j * 32);
            v16bf b1 = load_frag(Kp, DD, b * NN + m0 + 16, h * DKK + j * 32);
            c[0][0] = wmma_bf16(a0, b0, c[0][0]);
            c[0][1] = wmma_bf16(a0, b1, c[0][1]);
            c[1][0] = wmma_bf16(a1, b0, c[1][0]);
            c[1][1] = wmma_bf16(a1, b1, c[1][1]);
        }

        float vom[2];
#pragma unroll
        for (int mi = 0; mi < 2; ++mi)
            vom[mi] = vo[((size_t)b * HH + h) * NN + m0 + mi * 16 + nc];

#pragma unroll
        for (int ni = 0; ni < 2; ++ni)
#pragma unroll
            for (int r = 0; r < 8; ++r) {
                int row = n0 + ni * 16 + r + half * 8;
                size_t sidx = ((size_t)(b * HH + h) * NN + row) * 2;
                float mxr = stats[sidx + 0];
                float ivr = stats[sidx + 1];
#pragma unroll
                for (int mi = 0; mi < 2; ++mi)
                    acc[ni][mi][r] += __expf(c[ni][mi][r] * scale - mxr) * (vom[mi] * ivr);
            }
    }

#pragma unroll
    for (int mi = 0; mi < 2; ++mi) {
        int m = m0 + mi * 16 + nc;
        float me = mask_e[b * NN + m];
#pragma unroll
        for (int ni = 0; ni < 2; ++ni)
#pragma unroll
            for (int r = 0; r < 8; ++r) {
                int row = n0 + ni * 16 + r + half * 8;
                float v = (acc[ni][mi][r] + rb[b * NN + row]) * mask_d[b * NN + row] * me;
                out[((size_t)b * NN + row) * NN + m] = v;
            }
    }
}

// ---------------------------------------------------------------------------
extern "C" void kernel_launch(void* const* d_in, const int* in_sizes, int n_in,
                              void* d_out, int out_size, void* d_ws, size_t ws_size,
                              hipStream_t stream) {
    const float* input_d = (const float*)d_in[0];
    const float* input_e = (const float*)d_in[1];
    const float* mask_d  = (const float*)d_in[2];
    const float* mask_e  = (const float*)d_in[3];
    const float* W_Q     = (const float*)d_in[4];
    const float* W_K     = (const float*)d_in[5];
    const float* W_V     = (const float*)d_in[6];
    const float* W_O     = (const float*)d_in[7];
    float* out = (float*)d_out;

    char* ws = (char*)d_ws;
    size_t off = 0;
    auto carve = [&](size_t bytes) -> void* {
        void* p = ws + off;
        off += (bytes + 255) & ~(size_t)255;
        return p;
    };
    bf16*  WtQ   = (bf16*) carve((size_t)DD * DD * 2);            // 2 MB
    bf16*  WtK   = (bf16*) carve((size_t)DD * DD * 2);            // 2 MB
    bf16*  Xd    = (bf16*) carve((size_t)BB * NN * DD * 2);       // 8 MB
    bf16*  Xe    = (bf16*) carve((size_t)BB * NN * DD * 2);       // 8 MB
    bf16*  Qp    = (bf16*) carve((size_t)BB * NN * DD * 2);       // 8 MB
    bf16*  Kp    = (bf16*) carve((size_t)BB * NN * DD * 2);       // 8 MB
    float* wvo   = (float*)carve((size_t)HH * DD * 4);            // 32 KB
    float* vo    = (float*)carve((size_t)BB * HH * NN * 4);       // 128 KB
    float* rb    = (float*)carve((size_t)BB * NN * 4);            // 16 KB
    float* stats = (float*)carve((size_t)BB * HH * NN * 2 * 4);   // 256 KB

    k_transpose_w<<<1024, 256, 0, stream>>>(W_Q, WtQ);
    k_transpose_w<<<1024, 256, 0, stream>>>(W_K, WtK);
    k_wvo       <<<32,   256, 0, stream>>>(W_V, W_O, wvo);
    k_cvt       <<<4096, 256, 0, stream>>>(input_d, Xd);
    k_cvt       <<<4096, 256, 0, stream>>>(input_e, Xe);
    k_resbias   <<<4096, 256, 0, stream>>>(input_d, W_O, rb);
    k_vo        <<<4096, 256, 0, stream>>>(input_e, wvo, vo);
    k_gemm      <<<dim3(8, 32),    256, 0, stream>>>(Xd, WtQ, Qp);
    k_gemm      <<<dim3(8, 32),    256, 0, stream>>>(Xe, WtK, Kp);
    k_stats     <<<dim3(8, 32),    256, 0, stream>>>(Qp, Kp, stats);
    k_out       <<<dim3(8, 16, 4), 256, 0, stream>>>(Qp, Kp, stats, vo, rb,
                                                     mask_d, mask_e, out);
}
